// GATLayer_52845277610394
// MI455X (gfx1250) — compile-verified, matrix operand order checked by reading, not run on hardware
//
#include <hip/hip_runtime.h>
#include <hip/hip_bf16.h>
#include <math.h>

// ---------------- problem constants (from reference) ----------------
#define N_NODES 50000
#define N_EDGES 800000
#define IN_F    128
#define HEADS   8
#define OUT_F   32
#define HF      256      // HEADS*OUT_F
#define WCOLS   512      // proj cols (256) + skip cols (256)

typedef _Float16 v16h __attribute__((ext_vector_type(16)));
typedef _Float16 v8h  __attribute__((ext_vector_type(8)));
typedef float    v8f  __attribute__((ext_vector_type(8)));

// ---------------- fp32 -> fp16 conversions ----------------
__global__ void gat_cvt_x(const float* __restrict__ x, _Float16* __restrict__ xh) {
    int t = blockIdx.x * blockDim.x + threadIdx.x;
    if (t < N_NODES * IN_F) xh[t] = (_Float16)x[t];
}

// Build W^T (column-major weights): wt[n*128 + k] = W[k, n], n in [0,512)
// n < 256 -> W_proj column, else W_skip column (n-256)
__global__ void gat_cvt_w(const float* __restrict__ wp, const float* __restrict__ wsk,
                          _Float16* __restrict__ wt) {
    int t = blockIdx.x * blockDim.x + threadIdx.x;
    if (t < WCOLS * IN_F) {
        int n = t >> 7;          // 0..511
        int k = t & 127;         // 0..127
        float v = (n < HF) ? wp[k * HF + n] : wsk[k * HF + (n - HF)];
        wt[t] = (_Float16)v;
    }
}

// ---------------- zero denom + global-max scratch ----------------
__global__ void gat_init(float* __restrict__ denom, unsigned int* __restrict__ gmax) {
    int t = blockIdx.x * blockDim.x + threadIdx.x;
    if (t < N_NODES * HEADS) denom[t] = 0.0f;
    if (t == 0) *gmax = 0u;  // ordered-uint rep below all finite-float reps
}

// ---------------- fused GEMM: proj = x@W_proj (-> ws), skip = x@W_skip (-> d_out) ----
// grid.x = N/16 row tiles, 256 threads = 8 waves; each wave owns 4 of the 32
// 16-wide column tiles (512 output cols total). K = 128 = 4 steps of 32.
__global__ __launch_bounds__(256)
void gat_gemm_wmma(const _Float16* __restrict__ xh,   // [N,128] row-major
                   const _Float16* __restrict__ wt,   // [512,128] = W^T
                   float* __restrict__ proj,          // [N,256]
                   float* __restrict__ skip)          // [N,256]  (== d_out)
{
    const int lane  = threadIdx.x & 31;
    const int wave  = threadIdx.x >> 5;
    const int g     = lane >> 4;    // half-wave group (ISA K-packing)
    const int m     = lane & 15;    // row (A) / col (B,C,D) within tile
    const int row0  = blockIdx.x * 16;
    const int ncol0 = wave * 64;    // 4 contiguous 16-col tiles per wave

    v8f acc[4] = {v8f{}, v8f{}, v8f{}, v8f{}};

    #pragma unroll
    for (int k0 = 0; k0 < IN_F; k0 += 32) {
        // A fragment (16x32 f16): lane holds row row0+m, K in {8g..8g+7, 16+8g..23+8g}
        const _Float16* abase = xh + (size_t)(row0 + m) * IN_F + k0 + 8 * g;
        v8h alo = *(const v8h*)(abase);
        v8h ahi = *(const v8h*)(abase + 16);
        v16h a = __builtin_shufflevector(alo, ahi,
                 0,1,2,3,4,5,6,7,8,9,10,11,12,13,14,15);
        #pragma unroll
        for (int t = 0; t < 4; ++t) {
            // B fragment (32x16 f16): lane holds column ncol0+t*16+m of W (row of W^T)
            const _Float16* bbase = wt + (size_t)(ncol0 + t * 16 + m) * IN_F + k0 + 8 * g;
            v8h blo = *(const v8h*)(bbase);
            v8h bhi = *(const v8h*)(bbase + 16);
            v16h b = __builtin_shufflevector(blo, bhi,
                     0,1,2,3,4,5,6,7,8,9,10,11,12,13,14,15);
            acc[t] = __builtin_amdgcn_wmma_f32_16x16x32_f16(
                         false, a, false, b, (short)0, acc[t], false, false);
        }
    }

    // D layout: element r of v8f holds M = r + 8*g, N = lane&15
    #pragma unroll
    for (int t = 0; t < 4; ++t) {
        int cg = ncol0 + t * 16 + m;                 // global output column 0..511
        float* base = (cg < HF) ? (proj + cg) : (skip + (cg - HF));
        #pragma unroll
        for (int r = 0; r < 8; ++r) {
            int row = row0 + r + 8 * g;
            base[(size_t)row * HF] = acc[t][r];
        }
    }
}

// ---------------- per-node attention score dots: s_src/s_tgt [N,H] ----------------
// one block per node; warp w handles head h=w; 32-lane reduction over F
__global__ __launch_bounds__(256)
void gat_node_scores(const float* __restrict__ proj,
                     const float* __restrict__ ssw,   // score_src [H*F]
                     const float* __restrict__ stw,   // score_tgt [H*F]
                     float* __restrict__ s_src, float* __restrict__ s_tgt)
{
    int n = blockIdx.x;
    int h = threadIdx.x >> 5;
    int f = threadIdx.x & 31;
    float p = proj[(size_t)n * HF + h * OUT_F + f];
    float a = p * ssw[h * OUT_F + f];
    float b = p * stw[h * OUT_F + f];
    #pragma unroll
    for (int off = 16; off > 0; off >>= 1) {
        a += __shfl_xor(a, off, 32);
        b += __shfl_xor(b, off, 32);
    }
    if (f == 0) {
        s_src[n * HEADS + h] = a;
        s_tgt[n * HEADS + h] = b;
    }
}

// ---------------- edge scores + global max (ordered-uint atomicMax) ----------------
__global__ __launch_bounds__(256)
void gat_edge_scores(const long long* __restrict__ ei,
                     const float* __restrict__ s_src, const float* __restrict__ s_tgt,
                     float* __restrict__ scores, unsigned int* __restrict__ gmax)
{
    int t = blockIdx.x * blockDim.x + threadIdx.x;
    float sc = -3.0e38f;
    if (t < N_EDGES * HEADS) {
        int e = t >> 3, h = t & 7;
        int s = (int)ei[e];
        int d = (int)ei[N_EDGES + e];
        float v = s_src[s * HEADS + h] + s_tgt[d * HEADS + h];
        sc = (v > 0.0f) ? v : 0.2f * v;       // leaky_relu(0.2)
        scores[t] = sc;
    }
    float w = sc;
    #pragma unroll
    for (int off = 16; off > 0; off >>= 1) w = fmaxf(w, __shfl_xor(w, off, 32));
    if ((threadIdx.x & 31) == 0) {
        unsigned bits = __float_as_uint(w);
        unsigned rep  = (bits & 0x80000000u) ? ~bits : (bits | 0x80000000u);
        atomicMax(gmax, rep);
    }
}

// ---------------- exp(score - max) + segment-sum denominator ----------------
__global__ __launch_bounds__(256)
void gat_exp_denom(const long long* __restrict__ ei,
                   float* __restrict__ scores,          // in: scores, out: exp_scores
                   const unsigned int* __restrict__ gmax,
                   float* __restrict__ denom)
{
    int t = blockIdx.x * blockDim.x + threadIdx.x;
    if (t >= N_EDGES * HEADS) return;
    unsigned u = *gmax;
    unsigned bits = (u & 0x80000000u) ? (u ^ 0x80000000u) : ~u;
    float M = __uint_as_float(bits);
    int e = t >> 3, h = t & 7;
    float ex = expf(scores[t] - M);
    scores[t] = ex;
    int tgt = (int)ei[N_EDGES + e];
    atomicAdd(&denom[tgt * HEADS + h], ex);
}

// ---------------- weighted scatter: out[tgt] += proj[src] * attn --------------
// 256 threads = one (h,f) slot each; EPB edges per block; contiguous 1KB rows,
// all traffic resolves in L2 (proj + out both fit in 192MB L2).
#define EPB 4
__global__ __launch_bounds__(256)
void gat_scatter(const long long* __restrict__ ei,
                 const float* __restrict__ proj,
                 const float* __restrict__ expsc,
                 const float* __restrict__ denom,
                 float* __restrict__ out)
{
    const int h  = threadIdx.x >> 5;
    const int e0 = blockIdx.x * EPB;
    #pragma unroll
    for (int j = 0; j < EPB; ++j) {
        int e = e0 + j;
        int src = (int)ei[e];
        int tgt = (int)ei[N_EDGES + e];
        if (j + 1 < EPB) {
            int sn = (int)ei[e + 1];
            __builtin_prefetch(proj + (size_t)sn * HF + threadIdx.x, 0, 1);
        }
        float attn = expsc[e * HEADS + h] / (denom[tgt * HEADS + h] + 1e-16f);
        float v = proj[(size_t)src * HF + threadIdx.x] * attn;
        atomicAdd(out + (size_t)tgt * HF + threadIdx.x, v);
    }
}

// ---------------- finalize: out = elu(accum(+skip) + bias) ----------------
__global__ __launch_bounds__(256)
void gat_finalize(float* __restrict__ out, const float* __restrict__ bias)
{
    int t = blockIdx.x * blockDim.x + threadIdx.x;
    if (t >= N_NODES * HF) return;
    float v = out[t] + bias[t & (HF - 1)];
    out[t] = (v > 0.0f) ? v : (expf(v) - 1.0f);
}

// ================================================================
extern "C" void kernel_launch(void* const* d_in, const int* in_sizes, int n_in,
                              void* d_out, int out_size, void* d_ws, size_t ws_size,
                              hipStream_t stream) {
    (void)in_sizes; (void)n_in; (void)out_size; (void)ws_size;

    const float*     x    = (const float*)d_in[0];
    const long long* ei   = (const long long*)d_in[1];   // int64 [2, E]
    const float*     Wp   = (const float*)d_in[2];
    const float*     Wsk  = (const float*)d_in[3];
    const float*     ssw  = (const float*)d_in[4];
    const float*     stw  = (const float*)d_in[5];
    const float*     bias = (const float*)d_in[6];
    float*           out  = (float*)d_out;               // [N, 256] fp32

    // ---- workspace carve-out (256B aligned slabs) ----
    char*  ws  = (char*)d_ws;
    size_t off = 0;
    auto alloc = [&](size_t bytes) -> void* {
        void* p = ws + off;
        off = (off + bytes + 255) & ~(size_t)255;
        return p;
    };
    _Float16*     xh     = (_Float16*)    alloc((size_t)N_NODES * IN_F * 2);   // 12.8 MB
    _Float16*     wt     = (_Float16*)    alloc((size_t)WCOLS * IN_F * 2);     // 128 KB
    float*        proj   = (float*)       alloc((size_t)N_NODES * HF * 4);     // 51.2 MB
    float*        scores = (float*)       alloc((size_t)N_EDGES * HEADS * 4);  // 25.6 MB
    float*        denom  = (float*)       alloc((size_t)N_NODES * HEADS * 4);  // 1.6 MB
    float*        s_src  = (float*)       alloc((size_t)N_NODES * HEADS * 4);  // 1.6 MB
    float*        s_tgt  = (float*)       alloc((size_t)N_NODES * HEADS * 4);  // 1.6 MB
    unsigned int* gmax   = (unsigned int*)alloc(256);

    // 1) precision conversion + weight transpose
    gat_cvt_x<<<(N_NODES * IN_F + 255) / 256, 256, 0, stream>>>(x, xh);
    gat_cvt_w<<<(WCOLS * IN_F + 255) / 256, 256, 0, stream>>>(Wp, Wsk, wt);

    // 2) zero denom / gmax (d_out needs no zeroing: GEMM writes skip into it)
    gat_init<<<(N_NODES * HEADS + 255) / 256, 256, 0, stream>>>(denom, gmax);

    // 3) fused WMMA GEMM: proj -> ws, skip -> d_out (accumulator seed)
    gat_gemm_wmma<<<N_NODES / 16, 256, 0, stream>>>(xh, wt, proj, out);

    // 4) per-node score dots
    gat_node_scores<<<N_NODES, 256, 0, stream>>>(proj, ssw, stw, s_src, s_tgt);

    // 5) edge-wise leaky-relu scores + global max
    gat_edge_scores<<<(N_EDGES * HEADS + 255) / 256, 256, 0, stream>>>(
        ei, s_src, s_tgt, scores, gmax);

    // 6) exp(score - max) + softmax denominator (segment sum over tgt)
    gat_exp_denom<<<(N_EDGES * HEADS + 255) / 256, 256, 0, stream>>>(
        ei, scores, gmax, denom);

    // 7) weighted feature scatter into d_out (on top of skip seed)
    gat_scatter<<<N_EDGES / EPB, 256, 0, stream>>>(ei, proj, scores, denom, out);

    // 8) bias + ELU, in place on d_out
    gat_finalize<<<(N_NODES * HF + 255) / 256, 256, 0, stream>>>(out, bias);

    // 9) second tuple element: copy edge_index (int64 raw bytes) after `out`
    hipMemcpyAsync((char*)d_out + (size_t)N_NODES * HF * 4, (const void*)ei,
                   (size_t)2 * N_EDGES * 8, hipMemcpyDeviceToDevice, stream);
}
